// FeatureMemoryMappingModule_50173807951909
// MI455X (gfx1250) — matrix operand order repaired and checked by reading, not compiled, not run on hardware
//
#include <hip/hip_runtime.h>
#include <hip/hip_bf16.h>

// Problem constants (from reference)
#define N_ROWS   16384
#define DIM_IN   1024
#define DIM_H    512
#define MEM_ROWS 8192
#define RATE     0.5f

typedef __attribute__((ext_vector_type(16))) __bf16 v16bf;
typedef __attribute__((ext_vector_type(8)))  __bf16 v8bf;
typedef __attribute__((ext_vector_type(8)))  float  v8f;

// ---------------------------------------------------------------------------
// bf16x3 WMMA: (ah+al) x (bh+bl) ~= ah*bh + al*bh + ah*bl  (drop al*bl)
// ---------------------------------------------------------------------------
__device__ __forceinline__ v8f wmma_bf16x3(v16bf ah, v16bf al, v16bf bh, v16bf bl, v8f c) {
    c = __builtin_amdgcn_wmma_f32_16x16x32_bf16(false, ah, false, bh, (short)0, c, false, false);
    c = __builtin_amdgcn_wmma_f32_16x16x32_bf16(false, al, false, bh, (short)0, c, false, false);
    c = __builtin_amdgcn_wmma_f32_16x16x32_bf16(false, ah, false, bl, (short)0, c, false, false);
    return c;
}

// ---------------------------------------------------------------------------
// Fragment loaders (ISA 7.12.2 layouts, wave32)
// A 16x32 bf16: lane m = lane&15; kb = (lane>>4)*8; elem e: K = kb + e + (e>=8 ? 8 : 0)
//   -> two contiguous 16B chunks at K offsets {kb, kb+16}
// B 32x16 bf16 (B[k][n] = Wt[n][k], Wt row-major): lane n = col0+(lane&15);
//   kb = (lane>>4)*8; elem e: K = kb + e + (e>=8 ? 8 : 0)  -> chunks at {kb, kb+16}
// ---------------------------------------------------------------------------
__device__ __forceinline__ v16bf load_a_frag(const __bf16* __restrict__ A, int lda,
                                             int row0, int k0) {
    const int lane = threadIdx.x & 31;
    const int m    = lane & 15;
    const int kb   = (lane >> 4) * 8;
    const __bf16* base = A + (size_t)(row0 + m) * lda + k0 + kb;
    v8bf a0 = *(const v8bf*)(base);
    v8bf a1 = *(const v8bf*)(base + 16);
    v16bf f;
#pragma unroll
    for (int e = 0; e < 8; ++e) { f[e] = a0[e]; f[e + 8] = a1[e]; }
    return f;
}

__device__ __forceinline__ v16bf load_b_frag(const __bf16* __restrict__ Wt, int ldw,
                                             int col0, int k0) {
    const int lane = threadIdx.x & 31;
    const int n    = col0 + (lane & 15);
    const int kb   = (lane >> 4) * 8;
    const __bf16* base = Wt + (size_t)n * ldw + k0 + kb;
    v8bf b0 = *(const v8bf*)(base);
    v8bf b1 = *(const v8bf*)(base + 16);
    v16bf f;
#pragma unroll
    for (int e = 0; e < 8; ++e) { f[e] = b0[e]; f[e + 8] = b1[e]; }
    return f;
}

// A fragment with on-the-fly fp32 -> split-bf16 (used only for X, read once)
__device__ __forceinline__ void load_a_frag_f32(const float* __restrict__ A, int lda,
                                                int row0, int k0, v16bf& hi, v16bf& lo) {
    const int lane = threadIdx.x & 31;
    const int m    = lane & 15;
    const int kb   = (lane >> 4) * 8;
    const float* base = A + (size_t)(row0 + m) * lda + k0 + kb;
#pragma unroll
    for (int e = 0; e < 16; ++e) {
        const int off = (e < 8) ? e : (e + 8);
        float v  = base[off];
        __bf16 h = (__bf16)v;
        hi[e] = h;
        lo[e] = (__bf16)(v - (float)h);
    }
}

// ---------------------------------------------------------------------------
// Kernel: split fp32 array into hi/lo bf16 planes
// ---------------------------------------------------------------------------
__global__ void k_split(const float* __restrict__ src, __bf16* __restrict__ hi,
                        __bf16* __restrict__ lo, int n) {
    int i = blockIdx.x * blockDim.x + threadIdx.x;
    if (i < n) {
        float v  = src[i];
        __bf16 h = (__bf16)v;
        hi[i] = h;
        lo[i] = (__bf16)(v - (float)h);
    }
}

// ---------------------------------------------------------------------------
// Kernel: float4 copy (memory -> updated_memory init)
// ---------------------------------------------------------------------------
__global__ void k_copy4(const float4* __restrict__ src, float4* __restrict__ dst, int n4) {
    int i = blockIdx.x * blockDim.x + threadIdx.x;
    if (i < n4) dst[i] = src[i];
}

// ---------------------------------------------------------------------------
// Kernel: row squared norms (one wave per row)
// ---------------------------------------------------------------------------
__global__ void k_rownorm(const float* __restrict__ A, float* __restrict__ out,
                          int rows, int cols) {
    const int wave = threadIdx.x >> 5;
    const int lane = threadIdx.x & 31;
    const int row  = blockIdx.x * (blockDim.x >> 5) + wave;
    if (row >= rows) return;
    float s = 0.f;
    for (int c = lane; c < cols; c += 32) {
        float v = A[(size_t)row * cols + c];
        s += v * v;
    }
#pragma unroll
    for (int off = 16; off; off >>= 1) s += __shfl_xor(s, off, 32);
    if (lane == 0) out[row] = s;
}

// ---------------------------------------------------------------------------
// Kernel: projection  p = X @ Wp^T + bp, emitted as split-bf16 planes
// one block per 16-row tile; each wave owns a 16x64 output strip (4 col tiles)
// ---------------------------------------------------------------------------
__global__ void k_proj(const float* __restrict__ X,
                       const __bf16* __restrict__ Wphi, const __bf16* __restrict__ Wplo,
                       const float* __restrict__ bp,
                       __bf16* __restrict__ Phi, __bf16* __restrict__ Plo) {
    const int row0  = blockIdx.x * 16;
    const int wave  = threadIdx.x >> 5;
    const int lane  = threadIdx.x & 31;
    const int mbase = (lane >> 4) * 8;
    const int nloc  = lane & 15;

    const int col0 = wave * 64;  // 8 waves x 64 cols = 512
    v8f acc[4] = {};
    for (int k0 = 0; k0 < DIM_IN; k0 += 32) {
        // issue all B loads for this K-step first, then convert A, then WMMA
        v16bf bh[4], bl[4];
#pragma unroll
        for (int g = 0; g < 4; ++g) {
            bh[g] = load_b_frag(Wphi, DIM_IN, col0 + g * 16, k0);
            bl[g] = load_b_frag(Wplo, DIM_IN, col0 + g * 16, k0);
        }
        v16bf ah, al;
        load_a_frag_f32(X, DIM_IN, row0, k0, ah, al);
#pragma unroll
        for (int g = 0; g < 4; ++g)
            acc[g] = wmma_bf16x3(ah, al, bh[g], bl[g], acc[g]);
    }
#pragma unroll
    for (int g = 0; g < 4; ++g) {
        const int n = col0 + g * 16 + nloc;
        const float bias = bp[n];
#pragma unroll
        for (int r = 0; r < 8; ++r) {
            float v  = acc[g][r] + bias;
            __bf16 h = (__bf16)v;
            size_t o = (size_t)(row0 + mbase + r) * DIM_H + n;
            Phi[o] = h;
            Plo[o] = (__bf16)(v - (float)h);
        }
    }
}

// ---------------------------------------------------------------------------
// Kernel: distances + argmin + gather closest rows
// d2(i,j) ~ mm[j] - 2 * (p_i . m_j)   (||p||^2 dropped: row-constant)
// P hi/lo tile (shared by all 8 waves) staged in LDS; per K-step all 8 B
// fragment loads are issued up front, then 12 WMMAs.
// ---------------------------------------------------------------------------
__global__ void k_dist_argmin(const __bf16* __restrict__ Phi, const __bf16* __restrict__ Plo,
                              const __bf16* __restrict__ Mhi, const __bf16* __restrict__ Mlo,
                              const float* __restrict__ M, const float* __restrict__ mm,
                              int* __restrict__ idx_out, float* __restrict__ closest_out) {
    __shared__ __bf16 sAhi[16][DIM_H];   // 16 KB
    __shared__ __bf16 sAlo[16][DIM_H];   // 16 KB
    __shared__ float sD[8][16];
    __shared__ int   sI[8][16];
    __shared__ int   sidx[16];
    const int row0  = blockIdx.x * 16;
    const int wave  = threadIdx.x >> 5;
    const int lane  = threadIdx.x & 31;
    const int mbase = (lane >> 4) * 8;
    const int nloc  = lane & 15;

    // stage the shared A tile (16 x 512 hi/lo) into LDS, 16B chunks
    for (int e = threadIdx.x; e < 16 * DIM_H / 8; e += 256) {
        const int r = e >> 6, c8 = (e & 63) * 8;
        const size_t o = (size_t)(row0 + r) * DIM_H + c8;
        *(v8bf*)&sAhi[r][c8] = *(const v8bf*)&Phi[o];
        *(v8bf*)&sAlo[r][c8] = *(const v8bf*)&Plo[o];
    }
    __syncthreads();

    float bestD[8];
    int   bestI[8];
#pragma unroll
    for (int r = 0; r < 8; ++r) { bestD[r] = 3.4e38f; bestI[r] = 0; }

    for (int ctg = wave; ctg < MEM_ROWS / 64; ctg += 8) {
        const int col0 = ctg * 64;
        // prefetch next 64-column group of the memory planes (global_prefetch_b8)
        if (ctg + 8 < MEM_ROWS / 64) {
            __builtin_prefetch(Mhi + (size_t)(col0 + 512 + nloc) * DIM_H, 0, 0);
            __builtin_prefetch(Mlo + (size_t)(col0 + 512 + nloc) * DIM_H, 0, 0);
        }
        v8f acc[4] = {};
        for (int k0 = 0; k0 < DIM_H; k0 += 32) {
            v16bf bh[4], bl[4];
#pragma unroll
            for (int g = 0; g < 4; ++g) {
                bh[g] = load_b_frag(Mhi, DIM_H, col0 + g * 16, k0);
                bl[g] = load_b_frag(Mlo, DIM_H, col0 + g * 16, k0);
            }
            v16bf ah = load_a_frag(&sAhi[0][0], DIM_H, 0, k0);
            v16bf al = load_a_frag(&sAlo[0][0], DIM_H, 0, k0);
#pragma unroll
            for (int g = 0; g < 4; ++g)
                acc[g] = wmma_bf16x3(ah, al, bh[g], bl[g], acc[g]);
        }
#pragma unroll
        for (int g = 0; g < 4; ++g) {
            const int j = col0 + g * 16 + nloc;
            const float mmj = mm[j];
#pragma unroll
            for (int r = 0; r < 8; ++r) {
                float d = mmj - 2.0f * acc[g][r];
                if (d < bestD[r] || (d == bestD[r] && j < bestI[r])) { bestD[r] = d; bestI[r] = j; }
            }
        }
    }

    // butterfly argmin across the 16 lanes sharing the same row group
#pragma unroll
    for (int off = 1; off < 16; off <<= 1) {
#pragma unroll
        for (int r = 0; r < 8; ++r) {
            float od = __shfl_xor(bestD[r], off, 32);
            int   oi = __shfl_xor(bestI[r], off, 32);
            if (od < bestD[r] || (od == bestD[r] && oi < bestI[r])) { bestD[r] = od; bestI[r] = oi; }
        }
    }
    if (nloc == 0) {
#pragma unroll
        for (int r = 0; r < 8; ++r) { sD[wave][mbase + r] = bestD[r]; sI[wave][mbase + r] = bestI[r]; }
    }
    __syncthreads();
    if (threadIdx.x < 16) {
        float bd = sD[0][threadIdx.x];
        int   bi = sI[0][threadIdx.x];
#pragma unroll
        for (int w = 1; w < 8; ++w) {
            float od = sD[w][threadIdx.x];
            int   oi = sI[w][threadIdx.x];
            if (od < bd || (od == bd && oi < bi)) { bd = od; bi = oi; }
        }
        sidx[threadIdx.x] = bi;
        idx_out[row0 + threadIdx.x] = bi;
    }
    __syncthreads();
    // gather closest rows (whole block): 16 rows x 512 f32
    for (int e = threadIdx.x; e < 16 * DIM_H; e += 256) {
        const int r = e >> 9, c = e & (DIM_H - 1);
        closest_out[(size_t)(row0 + r) * DIM_H + c] = M[(size_t)sidx[r] * DIM_H + c];
    }
}

// ---------------------------------------------------------------------------
// Kernel: upd = (p - closest) @ Wu^T + bu; atomic scatter into updated memory
// q tile staged in LDS as split bf16; each wave owns one 16x64 strip.
// ---------------------------------------------------------------------------
__global__ void k_update(const __bf16* __restrict__ Phi, const __bf16* __restrict__ Plo,
                         const float* __restrict__ closest,
                         const __bf16* __restrict__ Wuhi, const __bf16* __restrict__ Wulo,
                         const float* __restrict__ bu,
                         const int* __restrict__ idx, float* __restrict__ Um) {
    __shared__ __bf16 qhi[16][DIM_H];
    __shared__ __bf16 qlo[16][DIM_H];
    __shared__ int    sidx[16];
    const int row0 = blockIdx.x * 16;
    if (threadIdx.x < 16) sidx[threadIdx.x] = idx[row0 + threadIdx.x];
    for (int e = threadIdx.x; e < 16 * DIM_H; e += 256) {
        const int r = e >> 9, c = e & (DIM_H - 1);
        const size_t o = (size_t)(row0 + r) * DIM_H + c;
        float v  = (float)Phi[o] + (float)Plo[o] - closest[o];
        __bf16 h = (__bf16)v;
        qhi[r][c] = h;
        qlo[r][c] = (__bf16)(v - (float)h);
    }
    __syncthreads();

    const int wave  = threadIdx.x >> 5;
    const int lane  = threadIdx.x & 31;
    const int mbase = (lane >> 4) * 8;
    const int nloc  = lane & 15;

    const int col0 = wave * 64;  // 8 waves x 64 cols = 512
    v8f acc[4] = {};
    for (int k0 = 0; k0 < DIM_H; k0 += 32) {
        v16bf bh[4], bl[4];
#pragma unroll
        for (int g = 0; g < 4; ++g) {
            bh[g] = load_b_frag(Wuhi, DIM_H, col0 + g * 16, k0);
            bl[g] = load_b_frag(Wulo, DIM_H, col0 + g * 16, k0);
        }
        v16bf ah = load_a_frag(&qhi[0][0], DIM_H, 0, k0);
        v16bf al = load_a_frag(&qlo[0][0], DIM_H, 0, k0);
#pragma unroll
        for (int g = 0; g < 4; ++g)
            acc[g] = wmma_bf16x3(ah, al, bh[g], bl[g], acc[g]);
    }
#pragma unroll
    for (int g = 0; g < 4; ++g) {
        const int n = col0 + g * 16 + nloc;
        const float bias = bu[n];
#pragma unroll
        for (int r = 0; r < 8; ++r) {
            float upd = acc[g][r] + bias;
            atomicAdd(&Um[(size_t)sidx[mbase + r] * DIM_H + n], RATE * upd);
        }
    }
}

// ---------------------------------------------------------------------------
// launch
// ---------------------------------------------------------------------------
extern "C" void kernel_launch(void* const* d_in, const int* in_sizes, int n_in,
                              void* d_out, int out_size, void* d_ws, size_t ws_size,
                              hipStream_t stream) {
    const float* X  = (const float*)d_in[0];  // [N, 1024]
    const float* M  = (const float*)d_in[1];  // [8192, 512]
    const float* Wp = (const float*)d_in[2];  // [512, 1024]
    const float* bp = (const float*)d_in[3];  // [512]
    const float* Wu = (const float*)d_in[4];  // [512, 512]
    const float* bu = (const float*)d_in[5];  // [512]

    float* closest_out = (float*)d_out;                          // [N, 512]
    float* updated     = closest_out + (size_t)N_ROWS * DIM_H;   // [8192, 512]

    // workspace layout (bf16 split planes + norms + indices), ~51 MB
    char* ws = (char*)d_ws;
    size_t off = 0;
    __bf16* Phi  = (__bf16*)(ws + off); off += (size_t)N_ROWS  * DIM_H * 2;
    __bf16* Plo  = (__bf16*)(ws + off); off += (size_t)N_ROWS  * DIM_H * 2;
    __bf16* Mhi  = (__bf16*)(ws + off); off += (size_t)MEM_ROWS * DIM_H * 2;
    __bf16* Mlo  = (__bf16*)(ws + off); off += (size_t)MEM_ROWS * DIM_H * 2;
    __bf16* Wphi = (__bf16*)(ws + off); off += (size_t)DIM_H * DIM_IN * 2;
    __bf16* Wplo = (__bf16*)(ws + off); off += (size_t)DIM_H * DIM_IN * 2;
    __bf16* Wuhi = (__bf16*)(ws + off); off += (size_t)DIM_H * DIM_H * 2;
    __bf16* Wulo = (__bf16*)(ws + off); off += (size_t)DIM_H * DIM_H * 2;
    float*  mm   = (float*) (ws + off); off += (size_t)MEM_ROWS * 4;
    int*    idx  = (int*)   (ws + off);

    // pre-split reused operands into hi/lo bf16 planes (all L2-resident)
    {
        int n;
        n = MEM_ROWS * DIM_H; k_split<<<(n + 255) / 256, 256, 0, stream>>>(M,  Mhi,  Mlo,  n);
        n = DIM_H * DIM_IN;   k_split<<<(n + 255) / 256, 256, 0, stream>>>(Wp, Wphi, Wplo, n);
        n = DIM_H * DIM_H;    k_split<<<(n + 255) / 256, 256, 0, stream>>>(Wu, Wuhi, Wulo, n);
    }
    // init updated_memory = memory
    {
        const int n4 = MEM_ROWS * DIM_H / 4;
        k_copy4<<<(n4 + 255) / 256, 256, 0, stream>>>((const float4*)M, (float4*)updated, n4);
    }
    // ||m_j||^2
    k_rownorm<<<MEM_ROWS / 4, 128, 0, stream>>>(M, mm, MEM_ROWS, DIM_H);
    // p = X @ Wp^T + bp  -> split-bf16 planes
    k_proj<<<N_ROWS / 16, 256, 0, stream>>>(X, Wphi, Wplo, bp, Phi, Plo);
    // argmin_j (||m_j||^2 - 2 p.m_j), gather closest
    k_dist_argmin<<<N_ROWS / 16, 256, 0, stream>>>(Phi, Plo, Mhi, Mlo, M, mm, idx, closest_out);
    // scatter-add RATE * ((p-closest) @ Wu^T + bu)
    k_update<<<N_ROWS / 16, 256, 0, stream>>>(Phi, Plo, closest_out, Wuhi, Wulo, bu, idx, updated);
}
